// AHGCSP_GCN_Layer_79061757985141
// MI455X (gfx1250) — matrix operand order repaired and checked by reading, not compiled, not run on hardware
//
#include <hip/hip_runtime.h>
#include <hip/hip_bf16.h>

typedef __attribute__((ext_vector_type(16))) __bf16 v16bf;
typedef __attribute__((ext_vector_type(8)))  __bf16 v8bf;
typedef __attribute__((ext_vector_type(8)))  float  v8f;

#define B_ 8
#define N_ 2048
#define F_ 64
#define U_ 64

// ---- prep: inputs [B,N,F] f32 -> X^T [B,F,N] bf16 (so WMMA B-tiles are contiguous) ----
__global__ void prep_xt(const float* __restrict__ x, __bf16* __restrict__ xt) {
    int idx = blockIdx.x * blockDim.x + threadIdx.x;
    if (idx >= B_ * N_ * F_) return;
    int f = idx % F_;
    int n = (idx / F_) % N_;
    int b = idx / (F_ * N_);
    xt[((size_t)b * F_ + f) * N_ + n] = (__bf16)x[idx];
}

// ---- prep: Wd [F,U] f32 -> Wd^T [U,F] bf16 ----
__global__ void prep_wdt(const float* __restrict__ wd, __bf16* __restrict__ wdt) {
    int idx = blockIdx.x * blockDim.x + threadIdx.x;
    if (idx >= F_ * U_) return;
    int u = idx % U_;
    int f = idx / U_;
    wdt[(size_t)u * F_ + f] = (__bf16)wd[idx];
}

// ---- fused: Fusion = DL*W0 + Geo*W1 + KL*W2 ; online softmax ; P@X ; Dense+tanh ----
// One wave32 per 16-row tile. Single streaming pass over DL/W (HBM-bound, ~536MB -> ~23us).
__global__ __launch_bounds__(32)
void fused_gcn(const float*  __restrict__ DL,
               const float*  __restrict__ W,
               const float*  __restrict__ Geo,
               const float*  __restrict__ KL,
               const __bf16* __restrict__ XT,
               const __bf16* __restrict__ WdT,
               const float*  __restrict__ bd,
               float*        __restrict__ out)
{
    // G1 restage tile, padded to 72 bf16/row (144B) -> conflict-free ds_load_b128
    __shared__ __align__(16) __bf16 lds_g1[16][72];

    const int lane = threadIdx.x;       // 0..31
    const int tile = blockIdx.x;        // 0..1023
    const int b    = tile >> 7;         // tile / (N/16)
    const int n0   = (tile & 127) << 4;

    const int hi  = lane >> 4;          // which 16-lane half
    const int row = lane & 15;          // A-layout: lane -> matrix row
    const int n   = n0 + row;
    const int klo = hi * 8;             // A-layout K base: {klo..klo+8, klo+16..klo+24}

    const float*  dl_row  = DL  + ((size_t)(b * N_ + n)) * N_;
    const float*  w_row   = W   + ((size_t)(b * N_ + n)) * N_ * 3;
    const float*  geo_row = Geo + (size_t)n * N_;
    const float*  kl_row  = KL  + (size_t)n * N_;
    const __bf16* xt_b    = XT  + (size_t)b * F_ * N_;

    v8f acc[4];
    #pragma unroll
    for (int t = 0; t < 4; ++t) acc[t] = (v8f){};
    float m_i = -1e30f;                 // running row max
    float l_i = 0.0f;                   // running row sum

    for (int m0 = 0; m0 < N_; m0 += 32) {
        // ---- prefetch the HBM-resident streams two chunks ahead (-> L2 hits) ----
        const int mpf = m0 + 64;
        if (mpf < N_) {
            const int pb = mpf + klo;
            __builtin_prefetch(dl_row + pb, 0, 0);
            __builtin_prefetch(w_row + (size_t)pb * 3, 0, 0);
            __builtin_prefetch(w_row + (size_t)pb * 3 + 32, 0, 0);
            __builtin_prefetch(w_row + (size_t)pb * 3 + 64, 0, 0);
        }

        // ---- fusion scores for this lane's 16 A-slots (two runs of 8 contiguous m) ----
        float fus[16];
        #pragma unroll
        for (int r = 0; r < 2; ++r) {
            const int mb = m0 + klo + r * 16;
            float dls[8], gs[8], ks[8], wf[24];
            *(float4*)(&dls[0]) = *(const float4*)(dl_row  + mb);
            *(float4*)(&dls[4]) = *(const float4*)(dl_row  + mb + 4);
            *(float4*)(&gs[0])  = *(const float4*)(geo_row + mb);
            *(float4*)(&gs[4])  = *(const float4*)(geo_row + mb + 4);
            *(float4*)(&ks[0])  = *(const float4*)(kl_row  + mb);
            *(float4*)(&ks[4])  = *(const float4*)(kl_row  + mb + 4);
            #pragma unroll
            for (int q = 0; q < 6; ++q)
                *(float4*)(&wf[q * 4]) = *(const float4*)(w_row + (size_t)mb * 3 + q * 4);
            #pragma unroll
            for (int e = 0; e < 8; ++e)
                fus[r * 8 + e] = dls[e] * wf[e * 3 + 0]
                               + gs[e]  * wf[e * 3 + 1]
                               + ks[e]  * wf[e * 3 + 2];
        }

        // ---- online softmax (lane + lane^16 jointly cover the 32-wide chunk row) ----
        float cmax = fus[0];
        #pragma unroll
        for (int i = 1; i < 16; ++i) cmax = fmaxf(cmax, fus[i]);
        cmax = fmaxf(cmax, __shfl_xor(cmax, 16));

        // Rescale only when some row's max actually grew (rare after warm-up):
        // wave-uniform branch skips 1 exp + 8 bpermutes + 32 muls per chunk.
        if (__ballot(cmax > m_i) != 0ull) {
            const float m_new = fmaxf(m_i, cmax);
            const float scale = __expf(m_i - m_new);
            l_i *= scale;
            #pragma unroll
            for (int j = 0; j < 8; ++j) {
                float s = __shfl(scale, j + hi * 8);
                #pragma unroll
                for (int t = 0; t < 4; ++t) acc[t][j] *= s;
            }
            m_i = m_new;
        }

        float psum = 0.0f;
        v16bf a;
        #pragma unroll
        for (int i = 0; i < 16; ++i) {
            float p = __expf(fus[i] - m_i);
            psum += p;
            a[i] = (__bf16)p;
        }
        psum += __shfl_xor(psum, 16);
        l_i += psum;

        // ---- P @ X : 4 WMMAs (F = 4 tiles of 16) ----
        // B-layout (32x16 bf16): lanes 0-15 hold K=0..7,16..23, lanes 16-31 hold K=8..15,24..31
        const int kb = m0 + hi * 8;
        #pragma unroll
        for (int t = 0; t < 4; ++t) {
            const __bf16* src = xt_b + (size_t)(t * 16 + row) * N_ + kb;
            v8bf blo = *(const v8bf*)(src);
            v8bf bhi = *(const v8bf*)(src + 16);
            v16bf bm = __builtin_shufflevector(blo, bhi,
                0, 1, 2, 3, 4, 5, 6, 7, 8, 9, 10, 11, 12, 13, 14, 15);
            acc[t] = __builtin_amdgcn_wmma_f32_16x16x32_bf16(
                false, a, false, bm, (short)0, acc[t], false, false);
        }
    }

    // ---- normalize by l and restage G1 tile to LDS as bf16 (A-major) ----
    const float inv = 1.0f / l_i;
    #pragma unroll
    for (int j = 0; j < 8; ++j) {
        float s = __shfl(inv, j + hi * 8);
        #pragma unroll
        for (int t = 0; t < 4; ++t)
            lds_g1[j + hi * 8][t * 16 + row] = (__bf16)(acc[t][j] * s);
    }
    __syncthreads();   // single-wave WG: orders LDS store->load

    // ---- Dense(64) with WMMA: G1[16x64] @ Wd[64x64], K split into 2 chunks of 32 ----
    v8f dacc[4];
    #pragma unroll
    for (int t = 0; t < 4; ++t) dacc[t] = (v8f){};
    #pragma unroll
    for (int kc = 0; kc < 2; ++kc) {
        const __bf16* arow = &lds_g1[row][kc * 32 + klo];
        v8bf a0 = *(const v8bf*)(arow);        // K = klo..klo+7
        v8bf a1 = *(const v8bf*)(arow + 16);   // K = klo+16..klo+23
        v16bf am = __builtin_shufflevector(a0, a1,
            0, 1, 2, 3, 4, 5, 6, 7, 8, 9, 10, 11, 12, 13, 14, 15);
        #pragma unroll
        for (int t = 0; t < 4; ++t) {
            const __bf16* wsrc = WdT + (size_t)(t * 16 + row) * F_ + kc * 32 + hi * 8;
            v8bf b0 = *(const v8bf*)(wsrc);
            v8bf b1 = *(const v8bf*)(wsrc + 16);
            v16bf bm = __builtin_shufflevector(b0, b1,
                0, 1, 2, 3, 4, 5, 6, 7, 8, 9, 10, 11, 12, 13, 14, 15);
            dacc[t] = __builtin_amdgcn_wmma_f32_16x16x32_bf16(
                false, am, false, bm, (short)0, dacc[t], false, false);
        }
    }

    // ---- bias + tanh + store (C-layout: lane -> col, VGPR j -> row) ----
    #pragma unroll
    for (int t = 0; t < 4; ++t) {
        const int u  = t * 16 + row;
        const float bu = bd[u];
        #pragma unroll
        for (int j = 0; j < 8; ++j) {
            float x = dacc[t][j] + bu;
            x = fminf(fmaxf(x, -15.0f), 15.0f);
            float e = __expf(2.0f * x);
            float y = (e - 1.0f) / (e + 1.0f);
            out[((size_t)(b * N_ + n0 + j + hi * 8)) * U_ + u] = y;
        }
    }
}

extern "C" void kernel_launch(void* const* d_in, const int* in_sizes, int n_in,
                              void* d_out, int out_size, void* d_ws, size_t ws_size,
                              hipStream_t stream) {
    const float* inputs = (const float*)d_in[0];   // [B,N,F]
    const float* DL     = (const float*)d_in[1];   // [B,N,N]
    const float* W      = (const float*)d_in[2];   // [B,N,N,3]
    const float* Geo    = (const float*)d_in[3];   // [N,N]
    const float* KL     = (const float*)d_in[4];   // [N,N]
    const float* Wd     = (const float*)d_in[5];   // [F,U]
    const float* bd     = (const float*)d_in[6];   // [U]

    __bf16* xt  = (__bf16*)d_ws;                       // B*F*N bf16 = 2 MB
    __bf16* wdt = xt + (size_t)B_ * F_ * N_;           // U*F bf16 = 8 KB

    prep_xt <<<(B_ * N_ * F_ + 255) / 256, 256, 0, stream>>>(inputs, xt);
    prep_wdt<<<(F_ * U_ + 255) / 256,     256, 0, stream>>>(Wd, wdt);
    fused_gcn<<<B_ * (N_ / 16), 32, 0, stream>>>(DL, W, Geo, KL, xt, wdt, bd,
                                                 (float*)d_out);
}
